// KPN_39788577030865
// MI455X (gfx1250) — matrix operand order repaired and checked by reading, not compiled, not run on hardware
//
#include <hip/hip_runtime.h>
#include <hip/hip_bf16.h>

#define BN 2
#define HH 384
#define WWID 384
#define NF 64
#define KKTOT 576          // 9 * 64
#define NPIX (BN * HH * WWID)
#define CORE_STRIDE 48

typedef __bf16 bf16_t;
typedef __bf16 v16bf __attribute__((ext_vector_type(16)));
typedef float  v8f   __attribute__((ext_vector_type(8)));

// ---------------------------------------------------------------------------
// First conv (2->64), direct VALU (only 18 MACs/out), NCHW fp32 -> NHWC bf16
// ---------------------------------------------------------------------------
__global__ __launch_bounds__(256) void conv_first_kernel(
    const float* __restrict__ in, const float* __restrict__ w,
    const float* __restrict__ bias, bf16_t* __restrict__ out)
{
    int idx = blockIdx.x * 256 + threadIdx.x;        // (pixel, oc)
    if (idx >= NPIX * NF) return;
    int oc = idx & 63;
    int p  = idx >> 6;
    int b  = p / (HH * WWID);
    int rem = p - b * HH * WWID;
    int y = rem / WWID;
    int x = rem - y * WWID;

    float s = bias[oc];
#pragma unroll
    for (int c = 0; c < 2; ++c) {
#pragma unroll
        for (int kh = 0; kh < 3; ++kh) {
            int yy = y + kh - 1;
            if (yy < 0 || yy >= HH) continue;
#pragma unroll
            for (int kw = 0; kw < 3; ++kw) {
                int xx = x + kw - 1;
                if (xx < 0 || xx >= WWID) continue;
                float a = in[(((size_t)b * 2 + c) * HH + yy) * WWID + xx];
                float wv = w[((oc * 2 + c) * 3 + kh) * 3 + kw];
                s += a * wv;
            }
        }
    }
    out[(size_t)p * NF + oc] = (bf16_t)s;
}

// ---------------------------------------------------------------------------
// Weight prep: OIHW fp32 (cout,64,3,3) -> bf16 WB[coutPad][576], k=(kh*3+kw)*64+c
// ---------------------------------------------------------------------------
__global__ __launch_bounds__(256) void prep_w_kernel(
    const float* __restrict__ w, bf16_t* __restrict__ wb,
    int cout_pad, int cout_real)
{
    int idx = blockIdx.x * 256 + threadIdx.x;
    if (idx >= cout_pad * KKTOT) return;
    int o = idx / KKTOT;
    int k = idx - o * KKTOT;
    int pos = k >> 6;
    int c   = k & 63;
    int kh = pos / 3, kw = pos - kh * 3;
    float v = (o < cout_real) ? w[(((size_t)o * NF + c) * 3 + kh) * 3 + kw] : 0.0f;
    wb[(size_t)o * KKTOT + k] = (bf16_t)v;
}

// ---------------------------------------------------------------------------
// Implicit-GEMM 3x3 conv, C_in=64, via v_wmma_f32_16x16x32_bf16.
// Block = 256 thr = 8 waves; each wave: 2 M-tiles (32 px) x NT*16 channels.
// Weights async-staged into LDS (global_load_async_to_lds_b128, ASYNCcnt).
// ---------------------------------------------------------------------------
template<int NT, bool RELU, bool RESID, bool ABS_F32>
__global__ __launch_bounds__(256) void conv_wmma_kernel(
    const bf16_t* __restrict__ act,    // NHWC bf16 [B][H][W][64]
    const bf16_t* __restrict__ wb,     // [NT*16][576] bf16
    const float*  __restrict__ bias,
    const bf16_t* __restrict__ resid,  // NHWC bf16 (may alias out_bf)
    bf16_t* __restrict__ out_bf,       // NHWC bf16 [B][H][W][64]
    float*  __restrict__ out_f32,      // [B][H][W][CORE_STRIDE]
    int cout_real)
{
    extern __shared__ bf16_t lds_wb[];   // only LDS object -> LDS offset 0

    const int tid = threadIdx.x;
    // async stage of WB into LDS (memory -> LDS direct, tracked by ASYNCcnt)
    {
        const uint4* src = (const uint4*)wb;
        const int nvec = NT * 16 * KKTOT / 8;    // uint4 count
        for (int i = tid; i < nvec; i += 256) {
            unsigned lds_off = (unsigned)(i * 16);
            const uint4* gp = src + i;
            asm volatile("global_load_async_to_lds_b128 %0, %1, off"
                         :: "v"(lds_off), "v"(gp) : "memory");
        }
        asm volatile("s_wait_asynccnt 0" ::: "memory");
    }
    __syncthreads();

    const int wave = tid >> 5;
    const int lane = tid & 31;
    const int half = lane >> 4;       // 0/1
    const int l16  = lane & 15;

    const int pbase = blockIdx.x * 256 + wave * 32;  // 2 tiles of 16 px
    const int b   = pbase / (HH * WWID);
    const int rem = pbase - b * HH * WWID;
    const int y   = rem / WWID;
    const int x0  = rem - y * WWID;                  // 384 % 32 == 0: one row

    // prefetch the three activation rows this wave will traverse
#pragma unroll
    for (int kh = 0; kh < 3; ++kh) {
        int yy = y + kh - 1;
        if (yy >= 0 && yy < HH)
            __builtin_prefetch(act + (((size_t)b * HH + yy) * WWID + x0) * NF, 0, 1);
    }

    v8f acc[2][NT];
#pragma unroll
    for (int t = 0; t < 2; ++t)
#pragma unroll
        for (int n = 0; n < NT; ++n) acc[t][n] = (v8f)0.0f;

    for (int pos = 0; pos < 9; ++pos) {
        const int kh = pos / 3, kw = pos - kh * 3;
        const int yy = y + kh - 1;
        const bool vy = (yy >= 0) && (yy < HH);

#pragma unroll
        for (int ci = 0; ci < 2; ++ci) {             // c0 = 0, 32
            const int k0 = (pos * 2 + ci) * 32;

            union { v16bf v; uint4 q[2]; } A[2];
#pragma unroll
            for (int t = 0; t < 2; ++t) {
                const int xx = x0 + t * 16 + l16 + kw - 1;
                if (vy && xx >= 0 && xx < WWID) {
                    const uint4* ap = (const uint4*)(act +
                        ((((size_t)b * HH + yy) * WWID + xx) * NF + half * 8));
                    A[t].q[0] = ap[ci * 4];          // K = c0 + half*8 .. +7
                    A[t].q[1] = ap[ci * 4 + 2];      // K += 16
                } else {
                    uint4 z; z.x = z.y = z.z = z.w = 0u;
                    A[t].q[0] = z; A[t].q[1] = z;
                }
            }

#pragma unroll
            for (int n = 0; n < NT; ++n) {
                union { v16bf v; uint4 q[2]; } Bf;
                const uint4* bp = (const uint4*)(lds_wb +
                    ((size_t)(n * 16 + l16) * KKTOT + k0 + half * 8));
                Bf.q[0] = bp[0];
                Bf.q[1] = bp[2];
                acc[0][n] = __builtin_amdgcn_wmma_f32_16x16x32_bf16(
                    false, A[0].v, false, Bf.v, (short)0, acc[0][n], false, false);
                acc[1][n] = __builtin_amdgcn_wmma_f32_16x16x32_bf16(
                    false, A[1].v, false, Bf.v, (short)0, acc[1][n], false, false);
            }
        }
    }

    // Epilogue: D layout -> lane(l16)=N, VGPR v -> M = v + half*8
#pragma unroll
    for (int t = 0; t < 2; ++t) {
#pragma unroll
        for (int n = 0; n < NT; ++n) {
            const int ch = n * 16 + l16;
            const float bv = (ch < cout_real) ? bias[ch] : 0.0f;
#pragma unroll
            for (int v = 0; v < 8; ++v) {
                const int mm = v + half * 8;
                const size_t pix = (size_t)pbase + t * 16 + mm;
                float val = acc[t][n][v] + bv;
                if (ABS_F32) {
                    out_f32[pix * CORE_STRIDE + ch] = fabsf(val);
                } else {
                    if (RELU)  val = fmaxf(val, 0.0f);
                    if (RESID) val += (float)resid[pix * NF + ch];
                    out_bf[pix * NF + ch] = (bf16_t)val;
                }
            }
        }
    }
}

// ---------------------------------------------------------------------------
// KPN blend: per pixel, 7 radial sections, softmax over KxK logits built from
// |core| channels via radial interpolation, blended with shifted input frame.
// ---------------------------------------------------------------------------
__device__ __forceinline__ float kpn_logit(const float* __restrict__ cp,
                                           int off, int i, int j, int mid)
{
    const int di = i - mid, dj = j - mid;
    const int d2 = di * di + dj * dj;
    if (d2 > mid * mid) return 0.0f;   // outside radius -> zero logit
    const float d = sqrtf((float)d2);
    const int lo = (int)d;             // floor (d >= 0)
    const float frac = d - (float)lo;
    float v = cp[off + lo] * (1.0f - frac);
    if (frac > 0.0f) v += cp[off + lo + 1] * frac;
    return v;
}

__global__ __launch_bounds__(256) void kpn_kernel(
    const float* __restrict__ core,   // [B][H][W][CORE_STRIDE], already |.|
    const float* __restrict__ frames, // [B][H][W] fp32
    float* __restrict__ out)          // [B][1][H][W]
{
    const int p = blockIdx.x * 256 + threadIdx.x;
    if (p >= NPIX) return;
    const int b = p / (HH * WWID);
    const int rem = p - b * HH * WWID;
    const int y = rem / WWID;
    const int x = rem - y * WWID;
    const float* cp = core + (size_t)p * CORE_STRIDE;

    float acc = 0.0f;
    int off = 0;
    for (int w = 2; w <= 8; ++w) {
        const int mid = w - 1;
        const int K = 2 * w - 1;
        // pass 1: max logit for stable softmax
        float mx = -1e30f;
        for (int i = 0; i < K; ++i)
            for (int j = 0; j < K; ++j)
                mx = fmaxf(mx, kpn_logit(cp, off, i, j, mid));
        // pass 2: softmax-weighted blend of shifted frame
        float den = 0.0f, num = 0.0f;
        for (int i = 0; i < K; ++i) {
            const int yy = y + i - mid;
            for (int j = 0; j < K; ++j) {
                const float e = __expf(kpn_logit(cp, off, i, j, mid) - mx);
                den += e;
                const int xx = x + j - mid;
                float patch = 0.0f;
                if (yy >= 0 && yy < HH && xx >= 0 && xx < WWID)
                    patch = frames[((size_t)b * HH + yy) * WWID + xx];
                num += e * patch;
            }
        }
        acc += num / den;
        off += w;
    }
    out[p] = acc;
}

// ---------------------------------------------------------------------------
extern "C" void kernel_launch(void* const* d_in, const int* in_sizes, int n_in,
                              void* d_out, int out_size, void* d_ws, size_t ws_size,
                              hipStream_t stream)
{
    const float* data_with_est = (const float*)d_in[0];
    const float* data          = (const float*)d_in[1];
    const float* w_first = (const float*)d_in[2];
    const float* b_first = (const float*)d_in[3];
    const float* wA[3] = { (const float*)d_in[4],  (const float*)d_in[8],  (const float*)d_in[12] };
    const float* bA[3] = { (const float*)d_in[5],  (const float*)d_in[9],  (const float*)d_in[13] };
    const float* wB[3] = { (const float*)d_in[6],  (const float*)d_in[10], (const float*)d_in[14] };
    const float* bB[3] = { (const float*)d_in[7],  (const float*)d_in[11], (const float*)d_in[15] };
    const float* w_out = (const float*)d_in[16];
    const float* b_out = (const float*)d_in[17];

    // workspace carve-up
    const size_t szX    = (size_t)NPIX * NF * sizeof(bf16_t);          // 37,748,736
    const size_t szCore = (size_t)NPIX * CORE_STRIDE * sizeof(float);  // 56,623,104
    char* ws = (char*)d_ws;
    bf16_t* xA   = (bf16_t*)(ws);
    bf16_t* xB   = (bf16_t*)(ws + szX);
    float*  core = (float*)(ws + 2 * szX);
    bf16_t* WB   = (bf16_t*)(ws + 2 * szX + szCore);                   // 73,728 B max

    const int convBlocks = NPIX / 256;            // 1152: 8 waves x 32 px
    const size_t lds64 = (size_t)64 * KKTOT * sizeof(bf16_t);   // 73,728 B
    const size_t lds48 = (size_t)48 * KKTOT * sizeof(bf16_t);   // 55,296 B

    // first conv: 2 -> 64 (VALU), NCHW fp32 -> NHWC bf16
    conv_first_kernel<<<(NPIX * NF + 255) / 256, 256, 0, stream>>>(
        data_with_est, w_first, b_first, xA);

    // 3 resblocks: xA -> (relu conv) -> xB -> (conv + resid) -> xA (in place)
    for (int r = 0; r < 3; ++r) {
        prep_w_kernel<<<(64 * KKTOT + 255) / 256, 256, 0, stream>>>(wA[r], WB, 64, 64);
        conv_wmma_kernel<4, true, false, false><<<convBlocks, 256, lds64, stream>>>(
            xA, WB, bA[r], nullptr, xB, nullptr, 64);
        prep_w_kernel<<<(64 * KKTOT + 255) / 256, 256, 0, stream>>>(wB[r], WB, 64, 64);
        conv_wmma_kernel<4, false, true, false><<<convBlocks, 256, lds64, stream>>>(
            xB, WB, bB[r], xA, xA, nullptr, 64);
    }

    // out conv: 64 -> 35 (padded to 48), epilogue |.| -> f32 core
    prep_w_kernel<<<(48 * KKTOT + 255) / 256, 256, 0, stream>>>(w_out, WB, 48, 35);
    conv_wmma_kernel<3, false, false, true><<<convBlocks, 256, lds48, stream>>>(
        xA, WB, b_out, nullptr, nullptr, core, 35);

    // radial-kernel softmax blend
    kpn_kernel<<<(NPIX + 255) / 256, 256, 0, stream>>>(core, data, (float*)d_out);
}
